// Multihead_2551210574158
// MI455X (gfx1250) — matrix operand order repaired and checked by reading, not compiled, not run on hardware
//
#include <hip/hip_runtime.h>
#include <hip/hip_bf16.h>

typedef __attribute__((ext_vector_type(16))) _Float16     v16h;
typedef __attribute__((ext_vector_type(8)))  float        v8f;
typedef __attribute__((ext_vector_type(4)))  unsigned int u32x4;
typedef __attribute__((ext_vector_type(8)))  unsigned int u32x8;

#define B_  4
#define N_  2048
#define D_  512
#define H_  8
#define E_  64

static __device__ __forceinline__ v8f wmma16(v16h a, v16h b, v8f c) {
  // D = A(16x32 f16) * B(32x16 f16) + C(16x16 f32)
  return __builtin_amdgcn_wmma_f32_16x16x32_f16(false, a, false, b, (short)0, c,
                                                false, false);
}

// A-operand (16x32, f16) from an f32 row-major matrix, converting to f16.
// lanes 0-15 = row lane,   halves 0-7 -> K 0..7,  halves 8-15 -> K 16..23
// lanes 16-31 = row lane-16, halves 0-7 -> K 8..15, halves 8-15 -> K 24..31
static __device__ __forceinline__ v16h load_a_f32(const float* src, int ld, int lane) {
  const int m  = lane & 15;
  const int kh = (lane >> 4) << 3;
  const float* p = src + (size_t)m * ld;
  v16h a;
#pragma unroll
  for (int i = 0; i < 8; ++i) {
    a[i]     = (_Float16)p[kh + i];
    a[8 + i] = (_Float16)p[16 + kh + i];
  }
  return a;
}

static __device__ __forceinline__ v16h load_a_f16(const _Float16* src, int ld, int lane) {
  const int m  = lane & 15;
  const int kh = (lane >> 4) << 3;
  const _Float16* p = src + (size_t)m * ld;
  v16h a;
#pragma unroll
  for (int i = 0; i < 8; ++i) {
    a[i]     = p[kh + i];
    a[8 + i] = p[16 + kh + i];
  }
  return a;
}

// B-operand (32x16, f16). Column n of B must be a contiguous row of `src`.
// lanes 0-15: col n = lane, halves 0-7 -> K 0..7, halves 8-15 -> K 16..23;
// lanes 16-31: col n = lane-16, halves 0-7 -> K 8..15, halves 8-15 -> K 24..31.
static __device__ __forceinline__ v16h load_b_f16(const _Float16* src, int ld, int lane) {
  const int n  = lane & 15;
  const int kb = (lane >> 4) << 3;
  const _Float16* p = src + (size_t)n * ld;
  v16h b;
#pragma unroll
  for (int i = 0; i < 8; ++i) {
    b[i]     = p[kb + i];
    b[8 + i] = p[16 + kb + i];
  }
  return b;
}

// ---- Tensor Data Mover: 2D f16 tile Global -> LDS (wave-uniform operands) ----
// D# group0: count=1 | lds_addr | global_addr | type=2
// D# group1: data_size=2B, tensor dims == tile dims (tiles always in-bounds)
static __device__ __forceinline__ void tdm_load_2d_f16(unsigned lds_addr,
                                                       const void* gptr,
                                                       unsigned tile0, unsigned tile1,
                                                       unsigned stride0_elems) {
  unsigned long long ga = (unsigned long long)(size_t)gptr;
  u32x4 g0;
  g0[0] = 1u;                                               // count=1 (valid user D#)
  g0[1] = lds_addr;                                         // lds_addr[31:0]
  g0[2] = (unsigned)ga;                                     // global_addr[31:0]
  g0[3] = ((unsigned)(ga >> 32) & 0x01FFFFFFu) | 0x80000000u; // addr[56:32] | type=2
  u32x8 g1;
  g1[0] = 1u << 16;                                         // data_size code 1 = 2B
  g1[1] = (tile0 & 0xFFFFu) << 16;                          // tensor_dim0[15:0]
  g1[2] = (tile1 & 0xFFFFu) << 16;                          // tensor_dim1[15:0]
  g1[3] = (tile0 & 0xFFFFu) << 16;                          // tile_dim0
  g1[4] = (tile1 & 0xFFFFu);                                // tile_dim1 (tile_dim2=0)
  g1[5] = stride0_elems;                                    // tensor_dim0_stride[31:0]
  g1[6] = 0u;
  g1[7] = 0u;
  asm volatile("tensor_load_to_lds %0, %1" :: "s"(g0), "s"(g1) : "memory");
}

// ---------------- weight prep ----------------
__global__ void transpose_w_kernel(const float* __restrict__ src,
                                   _Float16* __restrict__ dst) {
  int idx = blockIdx.x * blockDim.x + threadIdx.x;    // H*D*E = 262144 total
  int h = idx >> 15;
  int r = idx & 32767;
  int d = r >> 6;
  int e = r & 63;
  dst[(h << 15) + (e << 9) + d] = (_Float16)src[idx];
}

__global__ void convert_w_kernel(const float* __restrict__ src,
                                 _Float16* __restrict__ dst, int n) {
  int idx = blockIdx.x * blockDim.x + threadIdx.x;
  if (idx < n) dst[idx] = (_Float16)src[idx];
}

// ---------------- Q/K/V projections ----------------
__global__ void proj_kernel(const float* __restrict__ X,        // [B,N,D] f32
                            const _Float16* __restrict__ Wt,    // [H,E,D] f16
                            _Float16* __restrict__ dstRM,       // [B,H,N,E] (q/k)
                            _Float16* __restrict__ dstT,        // [B,H,E,N] (v)
                            int transposed) {
  const int lane = threadIdx.x;
  const int n0 = blockIdx.x << 4;
  const int b  = blockIdx.y >> 3;
  const int h  = blockIdx.y & 7;
  const float* Xb = X + ((size_t)(b * N_ + n0)) * D_;
  const _Float16* Wh = Wt + (size_t)h * E_ * D_;

  const v8f vzero = {0.f, 0.f, 0.f, 0.f, 0.f, 0.f, 0.f, 0.f};
  v8f acc[4];
#pragma unroll
  for (int c = 0; c < 4; ++c) acc[c] = vzero;

  for (int k0 = 0; k0 < D_; k0 += 32) {
    if (k0 + 32 < D_)
      __builtin_prefetch(Xb + (size_t)(lane & 15) * D_ + k0 + 32, 0, 1);
    v16h a = load_a_f32(Xb + k0, D_, lane);
#pragma unroll
    for (int c = 0; c < 4; ++c) {
      v16h bb = load_b_f16(Wh + (size_t)(c << 4) * D_ + k0, D_, lane);
      acc[c] = wmma16(a, bb, acc[c]);
    }
  }

  const int mrow = (lane >> 4) << 3;
  const int col  = lane & 15;
  const size_t bh = (size_t)(b * H_ + h);
  if (!transposed) {
    _Float16* dst = dstRM + bh * (size_t)N_ * E_;
#pragma unroll
    for (int c = 0; c < 4; ++c)
#pragma unroll
      for (int i = 0; i < 8; ++i)
        dst[(size_t)(n0 + mrow + i) * E_ + (c << 4) + col] = (_Float16)acc[c][i];
  } else {
    _Float16* dst = dstT + bh * (size_t)E_ * N_;
#pragma unroll
    for (int c = 0; c < 4; ++c)
#pragma unroll
      for (int i = 0; i < 8; ++i)
        dst[(size_t)((c << 4) + col) * N_ + n0 + mrow + i] = (_Float16)acc[c][i];
  }
}

// ---------------- flash attention with TDM double-buffered K/V staging -------
// Block = 4 waves; each wave owns one 16-query tile; K/V chunks shared via LDS.
__global__ void attn_kernel(const _Float16* __restrict__ Qh,   // [B,H,N,E]
                            const _Float16* __restrict__ Kh,   // [B,H,N,E]
                            const _Float16* __restrict__ Vt,   // [B,H,E,N]
                            _Float16* __restrict__ tmp) {      // [B,N,H*E]
  __shared__ _Float16 Kbuf[2][32 * E_];   // 2 x 4KB: 32 keys x 64
  __shared__ _Float16 Vbuf[2][E_ * 32];   // 2 x 4KB: 64 e-rows x 32 keys
  __shared__ _Float16 Pbuf[4][16 * 32];   // per-wave P staging

  const int lane = threadIdx.x & 31;
  const int wv   = threadIdx.x >> 5;
  const int nt0  = (blockIdx.x * 4 + wv) << 4;
  const int h = blockIdx.y;
  const int b = blockIdx.z;
  const size_t bh = (size_t)(b * H_ + h);
  const _Float16* Qb = Qh + bh * (size_t)N_ * E_ + (size_t)nt0 * E_;
  const _Float16* Kb = Kh + bh * (size_t)N_ * E_;
  const _Float16* Vb = Vt + bh * (size_t)E_ * N_;

  const unsigned kl0 = (unsigned)(size_t)&Kbuf[0][0];
  const unsigned kl1 = (unsigned)(size_t)&Kbuf[1][0];
  const unsigned vl0 = (unsigned)(size_t)&Vbuf[0][0];
  const unsigned vl1 = (unsigned)(size_t)&Vbuf[1][0];

  v16h qa0 = load_a_f16(Qb, E_, lane);        // K-dim 0..31 of E
  v16h qa1 = load_a_f16(Qb + 32, E_, lane);   // K-dim 32..63

  const v8f vzero = {0.f, 0.f, 0.f, 0.f, 0.f, 0.f, 0.f, 0.f};
  v8f acc[4];
#pragma unroll
  for (int c = 0; c < 4; ++c) acc[c] = vzero;
  float rm[8], rl[8];
#pragma unroll
  for (int i = 0; i < 8; ++i) { rm[i] = -1.0e30f; rl[i] = 0.f; }

  const float inv_hd = 1.0f / (float)E_;   // reference divides by head_dim
  const int mrow = (lane >> 4) << 3;
  const int col  = lane & 15;
  _Float16* Pw = &Pbuf[wv][0];

  // prologue: wave0 kicks off chunk 0 into buffer 0
  if (wv == 0) {
    tdm_load_2d_f16(kl0, Kb, E_, 32, E_);
    tdm_load_2d_f16(vl0, Vb, 32, E_, N_);
  }

  const int NCH = N_ / 32;
  for (int ch = 0; ch < NCH; ++ch) {
    const int buf = ch & 1;
    if (wv == 0) __builtin_amdgcn_s_wait_tensorcnt(0);  // chunk `ch` resident
    __syncthreads();                                    // publish to all waves
    if (wv == 0 && ch + 1 < NCH) {                      // prefetch chunk ch+1
      const int nn = (ch + 1) * 32;
      tdm_load_2d_f16(buf ? kl0 : kl1, Kb + (size_t)nn * E_, E_, 32, E_);
      tdm_load_2d_f16(buf ? vl0 : vl1, Vb + nn, 32, E_, N_);
    }

    const _Float16* Kl = &Kbuf[buf][0];
    const _Float16* Vl = &Vbuf[buf][0];

    // scores: S(16x32) = Q(16x64) . K_chunk^T
    v8f s0 = vzero, s1 = vzero;
    {
      v16h b00 = load_b_f16(Kl, E_, lane);
      v16h b01 = load_b_f16(Kl + 32, E_, lane);
      s0 = wmma16(qa0, b00, s0);
      s0 = wmma16(qa1, b01, s0);
      v16h b10 = load_b_f16(Kl + 16 * E_, E_, lane);
      v16h b11 = load_b_f16(Kl + 16 * E_ + 32, E_, lane);
      s1 = wmma16(qa0, b10, s1);
      s1 = wmma16(qa1, b11, s1);
    }
    // online softmax; each 16-lane half owns rows mrow..mrow+7 (one col/lane)
#pragma unroll
    for (int i = 0; i < 8; ++i) {
      float x0 = s0[i] * inv_hd;
      float x1 = s1[i] * inv_hd;
      float mx = fmaxf(x0, x1);
#pragma unroll
      for (int off = 1; off < 16; off <<= 1)
        mx = fmaxf(mx, __shfl_xor(mx, off, 16));
      float nm = fmaxf(rm[i], mx);
      float sc = __expf(rm[i] - nm);
      float p0 = __expf(x0 - nm);
      float p1 = __expf(x1 - nm);
      float rs = p0 + p1;
#pragma unroll
      for (int off = 1; off < 16; off <<= 1)
        rs += __shfl_xor(rs, off, 16);
      rl[i] = rl[i] * sc + rs;
      rm[i] = nm;
#pragma unroll
      for (int c = 0; c < 4; ++c) acc[c][i] *= sc;
      Pw[(mrow + i) * 32 + col]      = (_Float16)p0;
      Pw[(mrow + i) * 32 + 16 + col] = (_Float16)p1;
    }
    // per-wave LDS RAW fence (LDS is in-order per wave; wait drains stores)
    asm volatile("s_wait_dscnt 0x0" ::: "memory");
    // acc(16x64) += P(16x32) . V_chunk  (V staged transposed: [e][nc])
    v16h pa = load_a_f16(Pw, 32, lane);
#pragma unroll
    for (int c = 0; c < 4; ++c) {
      v16h vb = load_b_f16(Vl + (size_t)(c << 4) * 32, 32, lane);
      acc[c] = wmma16(pa, vb, acc[c]);
    }
    // next iteration's top barrier protects buffer reuse vs. the next TDM issue
  }

  // normalize, write head-concat f16 buffer [B, Nt, H*E]
  _Float16* orow = tmp + ((size_t)(b * N_ + nt0)) * D_ + h * E_;
#pragma unroll
  for (int i = 0; i < 8; ++i) {
    float inv = 1.0f / rl[i];
#pragma unroll
    for (int c = 0; c < 4; ++c)
      orow[(size_t)(mrow + i) * D_ + (c << 4) + col] = (_Float16)(acc[c][i] * inv);
  }
}

// ---------------- output projection: out = tmp @ Wout^T ----------------
__global__ void outproj_kernel(const _Float16* __restrict__ tmp,     // [B*N, D]
                               const _Float16* __restrict__ Wout_h,  // [D, D]
                               float* __restrict__ out) {            // [B*N, D]
  const int lane = threadIdx.x;
  const int r0 = blockIdx.x << 4;
  const int j0 = blockIdx.y << 6;
  const v8f vzero = {0.f, 0.f, 0.f, 0.f, 0.f, 0.f, 0.f, 0.f};
  v8f acc[4];
#pragma unroll
  for (int c = 0; c < 4; ++c) acc[c] = vzero;

  for (int k0 = 0; k0 < D_; k0 += 32) {
    v16h a = load_a_f16(tmp + (size_t)r0 * D_ + k0, D_, lane);
#pragma unroll
    for (int c = 0; c < 4; ++c) {
      v16h bb = load_b_f16(Wout_h + (size_t)(j0 + (c << 4)) * D_ + k0, D_, lane);
      acc[c] = wmma16(a, bb, acc[c]);
    }
  }
  const int mrow = (lane >> 4) << 3;
  const int col  = lane & 15;
#pragma unroll
  for (int c = 0; c < 4; ++c)
#pragma unroll
    for (int i = 0; i < 8; ++i)
      out[(size_t)(r0 + mrow + i) * D_ + j0 + (c << 4) + col] = acc[c][i];
}

extern "C" void kernel_launch(void* const* d_in, const int* in_sizes, int n_in,
                              void* d_out, int out_size, void* d_ws, size_t ws_size,
                              hipStream_t stream) {
  (void)in_sizes; (void)n_in; (void)out_size; (void)ws_size;
  const float* keys   = (const float*)d_in[0];
  const float* values = (const float*)d_in[1];
  const float* query  = (const float*)d_in[2];
  const float* Wk     = (const float*)d_in[3];
  const float* Wv     = (const float*)d_in[4];
  const float* Wq     = (const float*)d_in[5];
  const float* Wout   = (const float*)d_in[6];
  float* out = (float*)d_out;

  char* ws = (char*)d_ws;
  size_t off = 0;
  const size_t wsz  = (size_t)H_ * E_ * D_ * sizeof(_Float16);      // 512 KB
  const size_t qkv  = (size_t)B_ * H_ * N_ * E_ * sizeof(_Float16); // 8 MB
  _Float16* Wqt   = (_Float16*)(ws + off); off += wsz;
  _Float16* Wkt   = (_Float16*)(ws + off); off += wsz;
  _Float16* Wvt   = (_Float16*)(ws + off); off += wsz;
  _Float16* Wouth = (_Float16*)(ws + off); off += (size_t)D_ * D_ * sizeof(_Float16);
  _Float16* Qh    = (_Float16*)(ws + off); off += qkv;
  _Float16* Kh    = (_Float16*)(ws + off); off += qkv;
  _Float16* Vt    = (_Float16*)(ws + off); off += qkv;
  _Float16* tmp   = (_Float16*)(ws + off); off += qkv;              // [B,N,H*E]

  // weight prep
  transpose_w_kernel<<<1024, 256, 0, stream>>>(Wq, Wqt);
  transpose_w_kernel<<<1024, 256, 0, stream>>>(Wk, Wkt);
  transpose_w_kernel<<<1024, 256, 0, stream>>>(Wv, Wvt);
  convert_w_kernel<<<1024, 256, 0, stream>>>(Wout, Wouth, D_ * D_);

  dim3 blk(32);
  dim3 gproj(N_ / 16, B_ * H_);
  proj_kernel<<<gproj, blk, 0, stream>>>(query,  Wqt, Qh, nullptr, 0);
  proj_kernel<<<gproj, blk, 0, stream>>>(keys,   Wkt, Kh, nullptr, 0);
  proj_kernel<<<gproj, blk, 0, stream>>>(values, Wvt, nullptr, Vt, 1);

  dim3 gattn(N_ / 64, H_, B_);          // 4 query tiles per block
  attn_kernel<<<gattn, dim3(128), 0, stream>>>(Qh, Kh, Vt, tmp);

  dim3 gout((B_ * N_) / 16, D_ / 64);
  outproj_kernel<<<gout, blk, 0, stream>>>(tmp, Wouth, out);
}
